// HierAttNet_31155692765578
// MI455X (gfx1250) — matrix operand order repaired and checked by reading, not compiled, not run on hardware
//
#include <hip/hip_runtime.h>

// ---------------------------------------------------------------------------
// HierAttNet for MI455X (gfx1250, wave32).  bf16 WMMA 16x16x32, f32 accum.
// Sequential GRU chain uses the Tensor Data Mover to double-buffer gate
// inputs (gi) into LDS, overlapped with the WMMA phase.
// ---------------------------------------------------------------------------

typedef __bf16 bf16;
typedef __attribute__((ext_vector_type(16))) __bf16 v16bf;
typedef __attribute__((ext_vector_type(8)))  __bf16 v8bf;
typedef __attribute__((ext_vector_type(8)))  float  v8f;
typedef __attribute__((ext_vector_type(4)))  unsigned int u32x4;
typedef __attribute__((ext_vector_type(4)))  int i32x4;
typedef __attribute__((ext_vector_type(8)))  int i32x8;

#if defined(__HIP_DEVICE_COMPILE__) && __has_builtin(__builtin_amdgcn_tensor_load_to_lds)
#define HAS_TDM 1
#else
#define HAS_TDM 0
#endif

// ---- WMMA fragment helpers (contiguous-K order, same in A and B, 16-bit A 16x32 / B 32x16) -------
// A: lane L holds row M=L%16, 16 contiguous K at k0 + (L/16)*16.
//    Source row-major [M][K] bf16.
__device__ __forceinline__ v16bf load_frag_a(const bf16* X, int ld, int m0, int k0) {
  const int lane = threadIdx.x & 31;
  const bf16* p = X + (size_t)(m0 + (lane & 15)) * ld + (k0 + ((lane >> 4) << 4));
  v8bf lo = *(const v8bf*)p;
  v8bf hi = *(const v8bf*)(p + 8);
  return __builtin_shufflevector(lo, hi, 0,1,2,3,4,5,6,7,8,9,10,11,12,13,14,15);
}

// B: lane L holds N = n0 + L%16, 16 contiguous K at k0 + (L/16)*16.
//    Source row-major [N][K] bf16 (i.e. weight rows = output features).
__device__ __forceinline__ v16bf load_frag_b(const bf16* W, int ld, int n0, int k0) {
  const int lane = threadIdx.x & 31;
  return *(const v16bf*)(W + (size_t)(n0 + (lane & 15)) * ld + (k0 + ((lane >> 4) << 4)));
}

__device__ __forceinline__ v8f wmma_bf16(v16bf a, v16bf b, v8f c) {
  return __builtin_amdgcn_wmma_f32_16x16x32_bf16(false, a, false, b, (short)0, c, false, false);
}

__device__ __forceinline__ float sigm(float x) { return 1.f / (1.f + __expf(-x)); }

#if HAS_TDM
// 1-D TDM copy: 3072 x bf16 (6 KB) contiguous Global -> LDS.
// D# group0: count=1 | lds_addr | global_addr[56:0] | type=2 ("image").
// D# group1: data_size=1(2B); tensor_dim0=tile_dim0=3072; tensor_dim1=tile_dim1=1;
//            tensor_dim0_stride=3072; groups 2/3 zero (<=2-D tensor).
__device__ __forceinline__ void tdm_copy_gi(unsigned lds_off, const bf16* gsrc) {
  unsigned long long ga = (unsigned long long)(uintptr_t)gsrc;
  u32x4 g0 = { 1u, lds_off, (unsigned)ga,
               (unsigned)((ga >> 32) & 0x01ffffffu) | (2u << 30) };
  i32x8 g1 = { 1 << 16, 3072 << 16, 1 << 16, 3072 << 16, 1, 3072, 0, 0 };
  i32x4 z4 = { 0, 0, 0, 0 };
#if __clang_major__ >= 23
  i32x8 z8 = { 0, 0, 0, 0, 0, 0, 0, 0 };
  __builtin_amdgcn_tensor_load_to_lds(g0, g1, z4, z4, z8, 0);
#else
  __builtin_amdgcn_tensor_load_to_lds(g0, g1, z4, z4, 0);
#endif
}
#endif

// ---- conversion kernels ----------------------------------------------------
__global__ void cvt_bf16(const float* __restrict__ s, bf16* __restrict__ d, int n) {
  int i = blockIdx.x * 256 + threadIdx.x;
  if (i < n) d[i] = (bf16)s[i];
}

// WattT[n][k] = Watt[k][n]  (so B-fragments read contiguous K)
__global__ void cvt_bf16_T(const float* __restrict__ s, bf16* __restrict__ d, int dim) {
  int k = blockIdx.x * 16 + threadIdx.x;
  int n = blockIdx.y * 16 + threadIdx.y;
  d[(size_t)n * dim + k] = (bf16)s[(size_t)k * dim + n];
}

// ---- embedding gather -> bf16, rows ordered (s, t, b) ----------------------
__global__ void embed_gather(const int* __restrict__ ids, const float* __restrict__ tbl,
                             bf16* __restrict__ X) {
  int r = blockIdx.x;                       // r = (s*64 + t)*4 + b
  int b = r & 3, st = r >> 2, t = st & 63, s = st >> 6;
  int id = ids[(b * 256 + s) * 64 + t];     // input_ids [B,P,F,M,T]
  const float* src = tbl + (size_t)id * 256;
  bf16* dst = X + (size_t)r * 256;
  for (int i = threadIdx.x; i < 256; i += blockDim.x) dst[i] = (bf16)src[i];
}

// ---- gi = X @ Wih^T + bih ; output bf16 [R][768] ---------------------------
// grid (R/16, 768/64), block 32 (one wave computes a 16x64 tile).
__global__ __launch_bounds__(32) void gi_gemm(const bf16* __restrict__ X,
                                              const bf16* __restrict__ W,
                                              const float* __restrict__ bias,
                                              bf16* __restrict__ gi, int K) {
  const int m0 = blockIdx.x * 16, n0 = blockIdx.y * 64;
  v8f acc[4] = {};
  for (int k0 = 0; k0 < K; k0 += 32) {
    v16bf a = load_frag_a(X, K, m0, k0);
#pragma unroll
    for (int j = 0; j < 4; ++j)
      acc[j] = wmma_bf16(a, load_frag_b(W, K, n0 + j * 16, k0), acc[j]);
  }
  const int lane = threadIdx.x & 31, nn = lane & 15, mh = lane >> 4;
#pragma unroll
  for (int j = 0; j < 4; ++j) {
    int n = n0 + j * 16 + nn;
    float bn = bias[n];
#pragma unroll
    for (int v = 0; v < 8; ++v) {
      int m = m0 + v + 8 * mh;
      gi[(size_t)m * 768 + n] = (bf16)(acc[j][v] + bn);
    }
  }
}

// ---- sequential biGRU chain (one workgroup per direction) ------------------
// h (bf16 + f32) lives in LDS; Whh [768][256] bf16 streamed from WGP$/L2.
// 8 waves x 6 N-tiles.  gi gate inputs are double-buffered into LDS by the
// Tensor Data Mover one step ahead (wave 0 issues, s_wait_tensorcnt retires),
// fully overlapped with the WMMA phase.
__global__ __launch_bounds__(256) void rec_gru(const bf16* __restrict__ WhhF,
                                               const bf16* __restrict__ WhhB,
                                               const float* __restrict__ bhhF,
                                               const float* __restrict__ bhhB,
                                               const bf16* __restrict__ giF,
                                               const bf16* __restrict__ giB,
                                               bf16* __restrict__ y,
                                               int S_calls, int T) {
  const int dir = blockIdx.x;
  const bf16*  Whh = dir ? WhhB : WhhF;
  const float* bhh = dir ? bhhB : bhhF;
  const bf16*  gi  = dir ? giB  : giF;

  __shared__ bf16  h_bf[16][264];   // padded stride: conflict-free frag loads
  __shared__ float h_f [16][256];
  __shared__ float gh  [16][768];
#if HAS_TDM
  __shared__ bf16  gi_s[2][3072];   // TDM double buffer: 4 rows x 768 per step
#endif

  const int tid = threadIdx.x, wave = tid >> 5, lane = tid & 31;
  for (int i = tid; i < 16 * 264; i += 256) (&h_bf[0][0])[i] = (bf16)0.f;
  for (int i = tid; i < 16 * 256; i += 256) (&h_f[0][0])[i]  = 0.f;
  __syncthreads();

  const int steps = S_calls * T;
#if HAS_TDM
  const unsigned gi_s_base = (unsigned)(uintptr_t)&gi_s[0][0];
  if (wave == 0) {   // prime buffer 0 with step 0 (TDM ignores EXEC; 1 issue/wave)
    int t0 = dir ? (T - 1) : 0;
    tdm_copy_gi(gi_s_base, gi + (size_t)t0 * 4 * 768);
  }
#endif

  int s = 0, ti = 0;
  for (int p = 0; p < steps; ++p) {
    const int t = dir ? (T - 1 - ti) : ti;
    const size_t flat = (size_t)s * T + t;

    // gh = h @ Whh^T + bhh
    v16bf a[8];
#pragma unroll
    for (int kk = 0; kk < 8; ++kk) a[kk] = load_frag_a(&h_bf[0][0], 264, 0, kk * 32);
#pragma unroll
    for (int j = 0; j < 6; ++j) {
      const int nt = wave * 6 + j;
      v8f acc = {};
#pragma unroll
      for (int kk = 0; kk < 8; ++kk)
        acc = wmma_bf16(a[kk], load_frag_b(Whh, 256, nt * 16, kk * 32), acc);
      const int nn = nt * 16 + (lane & 15), mh = lane >> 4;
      const float bb = bhh[nn];
#pragma unroll
      for (int v = 0; v < 8; ++v) gh[v + 8 * mh][nn] = acc[v] + bb;
    }

#if HAS_TDM
    if (wave == 0) {
      if (p + 1 < steps) {          // issue next step's gi copy (other buffer)
        int ss = s, tt = ti + 1;
        if (tt == T) { tt = 0; ++ss; }
        int t2 = dir ? (T - 1 - tt) : tt;
        tdm_copy_gi(gi_s_base + (unsigned)(((p + 1) & 1) * (3072 * 2)),
                    gi + ((size_t)ss * T + t2) * 4 * 768);
        __builtin_amdgcn_s_wait_tensorcnt(1);  // retire current step's copy
      } else {
        __builtin_amdgcn_s_wait_tensorcnt(0);
      }
    }
#endif
    __syncthreads();

    // gate update: thread tid handles hidden index j=tid for batch rows 0..3
    {
      const int j = tid;
#if HAS_TDM
      const bf16* girow = &gi_s[p & 1][0];
#else
      const bf16* girow = gi + flat * 4 * 768;
#endif
#pragma unroll
      for (int m = 0; m < 4; ++m) {
        const bf16* g = girow + (size_t)m * 768;
        float r  = sigm((float)g[j]       + gh[m][j]);
        float z  = sigm((float)g[j + 256] + gh[m][j + 256]);
        float n_ = tanhf((float)g[j + 512] + r * gh[m][j + 512]);
        float hn = (1.f - z) * n_ + z * h_f[m][j];
        h_f[m][j]  = hn;
        h_bf[m][j] = (bf16)hn;
        y[(flat * 4 + m) * 512 + dir * 256 + j] = (bf16)hn;
      }
#if !HAS_TDM
      if (p + 1 < steps) {                       // prefetch next step's gi
        int s2 = s, ti2 = ti + 1;
        if (ti2 == T) { ti2 = 0; ++s2; }
        int t2 = dir ? (T - 1 - ti2) : ti2;
        const char* nx = (const char*)(gi + ((size_t)s2 * T + t2) * 4 * 768);
        if (tid < 48) __builtin_prefetch(nx + tid * 128, 0, 0);
      }
#endif
    }
    __syncthreads();
    if (++ti == T) { ti = 0; ++s; }
  }
}

// ---- attention: scores = tanh(y@Watt + batt)·ctx, masked softmax, Σ α·y ----
// grid = #calls, block 256.  Deterministic shuffle reduction (no atomics).
__global__ __launch_bounds__(256) void att_level(const bf16* __restrict__ y,
                                                 const bf16* __restrict__ WattT,
                                                 const float* __restrict__ batt,
                                                 const float* __restrict__ ctx,
                                                 const int* __restrict__ vlen,
                                                 int vstride, int S,
                                                 bf16* __restrict__ next_x,
                                                 float* __restrict__ final_out) {
  const int c = blockIdx.x, rows = S * 4;
  const bf16* yc = y + (size_t)c * rows * 512;
  __shared__ float sc[256];
  __shared__ float scW[8][256];      // per-wave partials -> deterministic sum
  const int tid = threadIdx.x, wave = tid >> 5, lane = tid & 31;
  for (int i = tid; i < 256; i += 256) sc[i] = 0.f;
  for (int i = tid; i < 8 * 256; i += 256) (&scW[0][0])[i] = 0.f;
  __syncthreads();

  const int Mt = rows >> 4;
  for (int idx = wave; idx < Mt * 32; idx += 8) {
    const int mt = idx >> 5, nt = idx & 31;
    v8f acc = {};
    for (int k0 = 0; k0 < 512; k0 += 32)
      acc = wmma_bf16(load_frag_a(yc, 512, mt * 16, k0),
                      load_frag_b(WattT, 512, nt * 16, k0), acc);
    const int n = nt * 16 + (lane & 15), mh = lane >> 4;
    const float bn = batt[n], cn = ctx[n];
#pragma unroll
    for (int v = 0; v < 8; ++v) {
      float val = tanhf(acc[v] + bn) * cn;
#pragma unroll
      for (int off = 1; off < 16; off <<= 1) val += __shfl_xor(val, off, 32);
      if ((lane & 15) == 0) scW[wave][mt * 16 + v + 8 * mh] += val;
    }
  }
  __syncthreads();
  if (tid < rows) {
    float a = 0.f;
#pragma unroll
    for (int w = 0; w < 8; ++w) a += scW[w][tid];
    sc[tid] = a;
  }
  __syncthreads();

  if (tid < 4) {                         // masked softmax over S per batch col
    const int b = tid;
    const int L = vlen ? vlen[b * vstride + c] : S;
    float mx = -1e30f;
    for (int q = 0; q < S; ++q) {
      float v = (q < L) ? sc[q * 4 + b] : -1e9f;
      sc[q * 4 + b] = v;
      mx = v > mx ? v : mx;
    }
    float sum = 0.f;
    for (int q = 0; q < S; ++q) { float e = __expf(sc[q * 4 + b] - mx); sc[q * 4 + b] = e; sum += e; }
    float inv = 1.f / sum;
    for (int q = 0; q < S; ++q) sc[q * 4 + b] *= inv;
  }
  __syncthreads();

#pragma unroll
  for (int u = 0; u < 8; ++u) {
    const int idx = u * 256 + tid;       // 2048 outputs = [4][512]
    const int b = idx >> 9, d = idx & 511;
    float sum = 0.f;
    for (int q = 0; q < S; ++q)
      sum += sc[q * 4 + b] * (float)yc[(size_t)(q * 4 + b) * 512 + d];
    if (next_x)    next_x[(size_t)(c * 4 + b) * 512 + d] = (bf16)sum;
    if (final_out) final_out[(size_t)b * 512 + d] = sum;
  }
}

// ---------------------------------------------------------------------------
extern "C" void kernel_launch(void* const* d_in, const int* in_sizes, int n_in,
                              void* d_out, int out_size, void* d_ws, size_t ws_size,
                              hipStream_t stream) {
  (void)in_sizes; (void)n_in; (void)out_size; (void)ws_size;
  const int*   ids  = (const int*)d_in[0];
  const int*   vlen = (const int*)d_in[1];
  const float* tbl  = (const float*)d_in[2];
  const float* tok[11];  for (int i = 0; i < 11; ++i) tok[i]  = (const float*)d_in[3 + i];
  const float* mth[11];  for (int i = 0; i < 11; ++i) mth[i]  = (const float*)d_in[14 + i];
  const float* fil[11];  for (int i = 0; i < 11; ++i) fil[i]  = (const float*)d_in[25 + i];
  const float* pkg[11];  for (int i = 0; i < 11; ++i) pkg[i]  = (const float*)d_in[36 + i];

  char* base = (char*)d_ws;
  size_t off = 0;
  auto abf = [&](size_t n) -> bf16* {
    off = (off + 255) & ~(size_t)255;
    bf16* p = (bf16*)(base + off);
    off += n * sizeof(bf16);
    return p;
  };

  // bf16 weight copies
  bf16 *wihF[4], *wihB[4], *whhF[4], *whhB[4], *wattT[4];
  const float* const* prm[4] = {tok, mth, fil, pkg};
  const int Kin[4] = {256, 512, 512, 512};
  for (int l = 0; l < 4; ++l) {
    wihF[l]  = abf((size_t)768 * Kin[l]);
    whhF[l]  = abf((size_t)768 * 256);
    wihB[l]  = abf((size_t)768 * Kin[l]);
    whhB[l]  = abf((size_t)768 * 256);
    wattT[l] = abf((size_t)512 * 512);
  }
  bf16* x_tok = abf((size_t)65536 * 256);
  bf16 *giF[4], *giB[4], *yb[4], *emb[4];
  const int R[4] = {65536, 1024, 128, 16};
  for (int l = 0; l < 4; ++l) {
    giF[l] = abf((size_t)R[l] * 768);
    giB[l] = abf((size_t)R[l] * 768);
    yb[l]  = abf((size_t)R[l] * 512);
    emb[l] = (l < 3) ? abf((size_t)(R[l + 1]) * 512) : nullptr;  // next level input
  }

  // ---- weight conversion ----
  auto cvt = [&](const float* s, bf16* d, int n) {
    cvt_bf16<<<(n + 255) / 256, 256, 0, stream>>>(s, d, n);
  };
  for (int l = 0; l < 4; ++l) {
    cvt(prm[l][0], wihF[l], 768 * Kin[l]);
    cvt(prm[l][1], whhF[l], 768 * 256);
    cvt(prm[l][4], wihB[l], 768 * Kin[l]);
    cvt(prm[l][5], whhB[l], 768 * 256);
    cvt_bf16_T<<<dim3(32, 32), dim3(16, 16), 0, stream>>>(prm[l][8], wattT[l], 512);
  }

  // ---- embedding gather ----
  embed_gather<<<65536, 64, 0, stream>>>(ids, tbl, x_tok);

  // ---- levels ----
  const int Scalls[4] = {256, 32, 4, 1};
  const int Tlen[4]   = {64, 8, 8, 4};
  const bf16* X = x_tok;
  for (int l = 0; l < 4; ++l) {
    dim3 g1(R[l] / 16, 12);  // 768/64 = 12 column tiles
    gi_gemm<<<g1, 32, 0, stream>>>(X, wihF[l], prm[l][2], giF[l], Kin[l]);
    gi_gemm<<<g1, 32, 0, stream>>>(X, wihB[l], prm[l][6], giB[l], Kin[l]);
    rec_gru<<<2, 256, 0, stream>>>(whhF[l], whhB[l], prm[l][3], prm[l][7],
                                   giF[l], giB[l], yb[l], Scalls[l], Tlen[l]);
    att_level<<<Scalls[l], 256, 0, stream>>>(
        yb[l], wattT[l], prm[l][9], prm[l][10],
        (l == 0) ? vlen : nullptr, (l == 0) ? 256 : 0, Tlen[l],
        (l < 3) ? emb[l] : nullptr, (l == 3) ? (float*)d_out : nullptr);
    X = emb[l];
  }
}
